// SparseResBlockWithObs_2078764172031
// MI455X (gfx1250) — matrix operand order repaired and checked by reading, not compiled, not run on hardware
//
#include <hip/hip_runtime.h>

typedef __attribute__((ext_vector_type(16))) _Float16 v16h;
typedef __attribute__((ext_vector_type(8)))  _Float16 v8h;
typedef __attribute__((ext_vector_type(8)))  float    v8f;

#define NPTS 500000
#define CCH  64
#define KNB  27
#define NBLK (NPTS / 32)                   // 15625 exactly (32-row blocks)
#define PKW_HALVES (KNB * 2 * 4 * 32 * 16) // 110592 packed f16 elems per layer
#define PKW_BYTES  (PKW_HALVES * 2)        // 221184 B  (< 320 KB WGP LDS)

union AFrag { v16h v; v8h h[2]; };

// ---------------- prep: fp32 -> fp16 feature conversion ----------------
__global__ void cvt_f32_to_f16(const float* __restrict__ src,
                               _Float16* __restrict__ dst, int n) {
  int i = blockIdx.x * blockDim.x + threadIdx.x;
  if (i < n) dst[i] = (_Float16)src[i];
}

// ---------------- prep: pack W[k] (27 x 64 x 64 row-major f32) into the
// WMMA B-fragment lane layout, flat index:
//   e = ((k*8 + c*4 + t) * 32 + lane) * 16 + h
//   B element = W[k][c*32 + (h>>3)*16 + (lane>>4)*8 + (h&7)][t*16 + (lane&15)]
__global__ void pack_w_kernel(const float* __restrict__ W,
                              _Float16* __restrict__ pk) {
  int e = blockIdx.x * blockDim.x + threadIdx.x;
  if (e >= PKW_HALVES) return;
  int h    = e & 15;
  int lane = (e >> 4) & 31;
  int t    = (e >> 9) & 3;
  int c    = (e >> 11) & 1;
  int k    = e >> 12;
  int kr   = c * 32 + (h >> 3) * 16 + (lane >> 4) * 8 + (h & 7);
  int col  = t * 16 + (lane & 15);
  pk[e] = (_Float16)W[(k * 64 + kr) * 64 + col];
}

// ---------------- main: gather + 27x GEMM(32x64x64) per 32-row block ----
// Each wave owns a 32x64 output block = two 16-row WMMA M-tiles, so every
// B fragment fetched from LDS feeds two v_wmma ops.
template <bool STAGE2>
__global__ __launch_bounds__(512) void sparse_conv_wmma(
    const _Float16* __restrict__ Asrc,   // N x 64 f16, row-major
    const _Float16* __restrict__ pkW,    // packed layer weights (global)
    const float*    __restrict__ bias,   // 64
    const float*    __restrict__ alpha,  // 1 (PReLU slope)
    const int*      __restrict__ nbr,    // N x 27
    const float*    __restrict__ resid,  // N x 64 f32 (stage 2 only)
    _Float16*       __restrict__ out_h,  // N x 64 f16 (stage 1 output)
    float*          __restrict__ out_f)  // N x 64 f32 (stage 2 output)
{
  extern __shared__ _Float16 sW[];       // 221184 B packed weights

  // cooperative copy of the whole layer's packed weights into LDS
  {
    const uint4* s = (const uint4*)pkW;
    uint4*       d = (uint4*)sW;
    for (int i = threadIdx.x; i < PKW_BYTES / 16; i += blockDim.x) d[i] = s[i];
  }
  __syncthreads();

  const int lane = threadIdx.x & 31;
  const int wave = threadIdx.x >> 5;
  const int nw   = blockDim.x >> 5;
  const int lm   = lane & 15;   // A row within tile / output column (mod 16)
  const int hi   = lane >> 4;   // half-wave selector
  const float al = alpha[0];

  v16h zero16;
#pragma unroll
  for (int i = 0; i < 16; ++i) zero16[i] = (_Float16)0.0f;

  float bv[4];
#pragma unroll
  for (int t = 0; t < 4; ++t) bv[t] = bias[t * 16 + lm];

  for (int blk = blockIdx.x * nw + wave; blk < NBLK; blk += gridDim.x * nw) {
    const int base = blk * 32;
    const int row0 = base + lm;        // gather row, M-tile 0
    const int row1 = base + 16 + lm;   // gather row, M-tile 1

    v8f acc[2][4];
#pragma unroll
    for (int t = 0; t < 4; ++t)
#pragma unroll
      for (int j = 0; j < 8; ++j) { acc[0][t][j] = bv[t]; acc[1][t][j] = bv[t]; }

    for (int k = 0; k < KNB; ++k) {
      int idx0 = nbr[row0 * KNB + k];
      int idx1 = nbr[row1 * KNB + k];
      const v8h* ap0 = (const v8h*)(Asrc + (size_t)(idx0 < 0 ? 0 : idx0) * CCH);
      const v8h* ap1 = (const v8h*)(Asrc + (size_t)(idx1 < 0 ? 0 : idx1) * CCH);

      // A fragments: [tile][k-chunk]; layout-exact 16B loads, ch = chunk*32
      AFrag a00, a01, a10, a11;
      a00.h[0] = ap0[hi];     a00.h[1] = ap0[2 + hi];
      a01.h[0] = ap0[4 + hi]; a01.h[1] = ap0[6 + hi];
      a10.h[0] = ap1[hi];     a10.h[1] = ap1[2 + hi];
      a11.h[0] = ap1[4 + hi]; a11.h[1] = ap1[6 + hi];
      if (idx0 < 0) { a00.v = zero16; a01.v = zero16; }
      if (idx1 < 0) { a10.v = zero16; a11.v = zero16; }

#pragma unroll
      for (int t = 0; t < 4; ++t) {
        AFrag b0, b1;
        const v8h* bp0 = (const v8h*)(sW + (size_t)(k * 8 + t)     * 512 + lane * 16);
        const v8h* bp1 = (const v8h*)(sW + (size_t)(k * 8 + 4 + t) * 512 + lane * 16);
        b0.h[0] = bp0[0]; b0.h[1] = bp0[1];
        b1.h[0] = bp1[0]; b1.h[1] = bp1[1];
        // chunk 0 (channels 0..31): both M-tiles share b0
        acc[0][t] = __builtin_amdgcn_wmma_f32_16x16x32_f16(
            false, a00.v, false, b0.v, (short)0, acc[0][t], false, false);
        acc[1][t] = __builtin_amdgcn_wmma_f32_16x16x32_f16(
            false, a10.v, false, b0.v, (short)0, acc[1][t], false, false);
        // chunk 1 (channels 32..63): both M-tiles share b1
        acc[0][t] = __builtin_amdgcn_wmma_f32_16x16x32_f16(
            false, a01.v, false, b1.v, (short)0, acc[0][t], false, false);
        acc[1][t] = __builtin_amdgcn_wmma_f32_16x16x32_f16(
            false, a11.v, false, b1.v, (short)0, acc[1][t], false, false);
      }
    }

    // epilogue: bias already in acc; residual + PReLU + store
#pragma unroll
    for (int tt = 0; tt < 2; ++tt) {
#pragma unroll
      for (int t = 0; t < 4; ++t) {
        const int col = t * 16 + lm;
#pragma unroll
        for (int j = 0; j < 8; ++j) {
          int r2 = base + tt * 16 + hi * 8 + j;   // C-fragment row mapping
          float v = acc[tt][t][j];
          if (STAGE2) {
            v += resid[(size_t)r2 * CCH + col];
            v = (v >= 0.0f) ? v : al * v;
            out_f[(size_t)r2 * CCH + col] = v;
          } else {
            v = (v >= 0.0f) ? v : al * v;
            out_h[(size_t)r2 * CCH + col] = (_Float16)v;
          }
        }
      }
    }
  }
}

extern "C" void kernel_launch(void* const* d_in, const int* in_sizes, int n_in,
                              void* d_out, int out_size, void* d_ws, size_t ws_size,
                              hipStream_t stream) {
  const float* feats = (const float*)d_in[0];
  const float* W1    = (const float*)d_in[1];
  const float* b1    = (const float*)d_in[2];
  const float* a1    = (const float*)d_in[3];
  const float* W2    = (const float*)d_in[4];
  const float* b2    = (const float*)d_in[5];
  const float* a2    = (const float*)d_in[6];
  const int*   nbr   = (const int*)d_in[7];
  float*       out   = (float*)d_out;

  // workspace layout: feats16 | xbuf16 | pkW1 | pkW2  (~128.4 MB total)
  _Float16* feats16 = (_Float16*)d_ws;
  _Float16* xbuf    = feats16 + (size_t)NPTS * CCH;
  _Float16* pkW1    = xbuf    + (size_t)NPTS * CCH;
  _Float16* pkW2    = pkW1    + PKW_HALVES;

  const int nconv = NPTS * CCH;
  cvt_f32_to_f16<<<(nconv + 511) / 512, 512, 0, stream>>>(feats, feats16, nconv);
  pack_w_kernel<<<(PKW_HALVES + 255) / 256, 256, 0, stream>>>(W1, pkW1);
  pack_w_kernel<<<(PKW_HALVES + 255) / 256, 256, 0, stream>>>(W2, pkW2);

  dim3 grid(512), block(512);
  // stage 1: x = prelu(conv(feats, W1) + b1, a1)  -> f16 xbuf
  sparse_conv_wmma<false><<<grid, block, PKW_BYTES, stream>>>(
      feats16, pkW1, b1, a1, nbr, nullptr, xbuf, nullptr);
  // stage 2: out = prelu(conv(x, W2) + b2 + feats, a2) -> f32 out
  sparse_conv_wmma<true><<<grid, block, PKW_BYTES, stream>>>(
      xbuf, pkW2, b2, a2, nbr, feats, nullptr, out);
}
